// Attention2d_27805618275195
// MI455X (gfx1250) — matrix operand order repaired and checked by reading, not compiled, not run on hardware
//
#include <hip/hip_runtime.h>
#include <hip/hip_bf16.h>
#include <math.h>

typedef _Float16 f16;
typedef __attribute__((ext_vector_type(2)))  _Float16 h2;
typedef __attribute__((ext_vector_type(16))) _Float16 v16h;
typedef __attribute__((ext_vector_type(8)))  float    v8f;

#define B_    1024
#define DIM_  384
#define N_    49
#define NP_   64
#define H_    8
#define KD_   32
#define D_    128
#define DH_   1024
#define OC_   1536              // fused q(256)+k(256)+v(1024) rows
#define MCOL_ (B_ * N_)         // 50176 spatial columns
#define NT4_  (MCOL_ / 64)      // 784 groups of 4 n-tiles
#define SCALE_ 0.1767766952966369f   // 32^-0.5

union F16Frag { v16h v; h2 p[8]; };

// ---- CDNA5 WMMA fragment gathers (per ISA 7.12.2 VGPR layouts, wave32) ----
// Both gathers are two contiguous 16B runs per lane -> compiler emits
// global_load_b128 / ds_load_b128 pairs.

// A-matrix 16x32 f16 from row-major [M][K] storage (K contiguous, stride ld halves).
__device__ __forceinline__ v16h load_frag_a(const f16* src, int ld) {
  const int lane = threadIdx.x & 31, lo = lane & 15, hi = lane >> 4;
  F16Frag f;
  const f16* row = src + lo * ld;
#pragma unroll
  for (int p = 0; p < 8; ++p) {
    const int kb = (p < 4 ? 2 * p : 16 + 2 * (p - 4)) + 8 * hi;
    f.p[p] = *(const h2*)(row + kb);
  }
  return f.v;
}

// B-matrix 32x16 f16 gathered from [N][K] storage (K contiguous, stride ld halves).
__device__ __forceinline__ v16h load_frag_b(const f16* src, int ld) {
  const int lane = threadIdx.x & 31, lo = lane & 15, hi = lane >> 4;
  F16Frag f;
  const f16* row = src + lo * ld + 8 * hi;
#pragma unroll
  for (int p = 0; p < 8; ++p) f.p[p] = *(const h2*)(row + (p < 4 ? 2 * p : 16 + 2 * (p - 4)));
  return f.v;
}

__device__ __forceinline__ v8f wmma16(v16h a, v16h b, v8f c) {
  return __builtin_amdgcn_wmma_f32_16x16x32_f16(false, a, false, b, (short)0, c,
                                                false, false);
}

// ---------------- K0: fold BN into weights, convert f16 ----------------
__global__ void __launch_bounds__(256) pack_weights(
    const float* qw, const float* qb, const float* qs, const float* qo,
    const float* kw, const float* kb, const float* ksc, const float* ko,
    const float* vw, const float* vb, const float* vsc, const float* vo,
    const float* pw, const float* pb, const float* psc, const float* po,
    f16* wqkv, float* qkvadd, f16* pwh, float* padd) {
  const int T1 = OC_ * DIM_, T2 = DIM_ * DH_;
  int i = blockIdx.x * blockDim.x + threadIdx.x;
  if (i < T1) {
    int o = i / DIM_, c = i % DIM_;
    float w, s;
    if (o < 256)      { w = qw[o * DIM_ + c];          s = qs[o]; }
    else if (o < 512) { int u = o - 256; w = kw[u * DIM_ + c]; s = ksc[u]; }
    else              { int u = o - 512; w = vw[u * DIM_ + c]; s = vsc[u]; }
    wqkv[i] = (f16)(w * s);
  } else if (i < T1 + T2) {
    int j = i - T1, o = j / DH_;
    pwh[j] = (f16)(pw[j] * psc[o]);
  } else if (i < T1 + T2 + OC_) {
    int o = i - T1 - T2;
    float bb, s, of;
    if (o < 256)      { bb = qb[o]; s = qs[o]; of = qo[o]; }
    else if (o < 512) { int u = o - 256; bb = kb[u]; s = ksc[u]; of = ko[u]; }
    else              { int u = o - 512; bb = vb[u]; s = vsc[u]; of = vo[u]; }
    qkvadd[o] = bb * s + of;
  } else if (i < T1 + T2 + OC_ + DIM_) {
    int o = i - T1 - T2 - OC_;
    padd[o] = pb[o] * psc[o] + po[o];
  }
}

// ---------------- K0b: x (B,C,49) -> xT f16 [m][c] ----------------
__global__ void __launch_bounds__(256) transpose_x(const float* __restrict__ x,
                                                   f16* __restrict__ xt) {
  long i = (long)blockIdx.x * blockDim.x + threadIdx.x;
  const long T = (long)MCOL_ * DIM_;
  if (i >= T) return;
  int c = (int)(i % DIM_);
  int m = (int)(i / DIM_);
  int b = m / N_, n = m % N_;
  xt[i] = (f16)x[((long)b * DIM_ + c) * N_ + n];
}

// ---------------- K0c: zero padded q/k/v region ----------------
__global__ void zero_u32(unsigned int* p, long n) {
  long i = (long)blockIdx.x * blockDim.x + threadIdx.x;
  long stride = (long)gridDim.x * blockDim.x;
  for (; i < n; i += stride) p[i] = 0u;
}

// ---------------- K1: fused QKV GEMM (1536 x 384 x 50176) ----------------
// K-loop unrolled by 2 with two named fragment register sets (no rotation
// moves); each half issues the next half's loads before its WMMAs so VMEM
// stays pipelined against the matrix unit.
__global__ void __launch_bounds__(256) qkv_gemm(
    const f16* __restrict__ wqkv, const f16* __restrict__ xt,
    const float* __restrict__ qkvadd, f16* __restrict__ qh,
    f16* __restrict__ kh, f16* __restrict__ vh) {
  const int wave = blockIdx.x * (blockDim.x >> 5) + (threadIdx.x >> 5);
  if (wave >= 96 * NT4_) return;  // wave-uniform
  const int mt = wave / NT4_, nt4 = wave % NT4_;
  const int m0 = mt * 16, n0 = nt4 * 64;
  const f16* aptr = wqkv + (long)m0 * DIM_;
  const f16* bptr = xt + (long)n0 * DIM_;

  v8f acc[4] = {};
  // ping set (k0), pong set (k0+32); DIM_ = 384 = 6 * 64 exactly.
  v16h a0 = load_frag_a(aptr, DIM_);
  v16h b0_0 = load_frag_b(bptr + 0 * 16 * DIM_, DIM_);
  v16h b0_1 = load_frag_b(bptr + 1 * 16 * DIM_, DIM_);
  v16h b0_2 = load_frag_b(bptr + 2 * 16 * DIM_, DIM_);
  v16h b0_3 = load_frag_b(bptr + 3 * 16 * DIM_, DIM_);

#pragma unroll 1
  for (int k0 = 0; k0 < DIM_; k0 += 64) {
    // pong loads (k0+32) issued before ping WMMAs
    const int kp = k0 + 32;
    v16h a1 = load_frag_a(aptr + kp, DIM_);
    v16h b1_0 = load_frag_b(bptr + 0 * 16 * DIM_ + kp, DIM_);
    v16h b1_1 = load_frag_b(bptr + 1 * 16 * DIM_ + kp, DIM_);
    v16h b1_2 = load_frag_b(bptr + 2 * 16 * DIM_ + kp, DIM_);
    v16h b1_3 = load_frag_b(bptr + 3 * 16 * DIM_ + kp, DIM_);
    __builtin_prefetch(aptr + k0 + 64, 0, 1);

    acc[0] = wmma16(a0, b0_0, acc[0]);
    acc[1] = wmma16(a0, b0_1, acc[1]);
    acc[2] = wmma16(a0, b0_2, acc[2]);
    acc[3] = wmma16(a0, b0_3, acc[3]);

    if (k0 + 64 < DIM_) {
      const int kn = k0 + 64;
      a0 = load_frag_a(aptr + kn, DIM_);
      b0_0 = load_frag_b(bptr + 0 * 16 * DIM_ + kn, DIM_);
      b0_1 = load_frag_b(bptr + 1 * 16 * DIM_ + kn, DIM_);
      b0_2 = load_frag_b(bptr + 2 * 16 * DIM_ + kn, DIM_);
      b0_3 = load_frag_b(bptr + 3 * 16 * DIM_ + kn, DIM_);
    }
    acc[0] = wmma16(a1, b1_0, acc[0]);
    acc[1] = wmma16(a1, b1_1, acc[1]);
    acc[2] = wmma16(a1, b1_2, acc[2]);
    acc[3] = wmma16(a1, b1_3, acc[3]);
  }

  const int lo = threadIdx.x & 15, hi = (threadIdx.x >> 4) & 1;
#pragma unroll
  for (int s = 0; s < 4; ++s) {
#pragma unroll
    for (int r = 0; r < 8; ++r) {
      const int o = m0 + r + 8 * hi;
      const int mc = n0 + s * 16 + lo;
      const int b = mc / N_, n = mc % N_;
      const f16 hv = (f16)(acc[s][r] + qkvadd[o]);
      if (o < 256) {
        int h = o >> 5, kd = o & 31;
        qh[(((long)b * H_ + h) * NP_ + n) * KD_ + kd] = hv;
      } else if (o < 512) {
        int u = o - 256, h = u >> 5, kd = u & 31;
        kh[(((long)b * H_ + h) * NP_ + n) * KD_ + kd] = hv;
      } else {
        int u = o - 512, h = u >> 7, d = u & 127;
        vh[(((long)b * H_ + h) * D_ + d) * NP_ + n] = hv;
      }
    }
  }
}

// ---------------- K2: attention (one block per batch, wave == head) -------
__global__ void __launch_bounds__(256) attn_kernel(
    const f16* __restrict__ qh, const f16* __restrict__ kh,
    const f16* __restrict__ vh, const float* __restrict__ th1w,
    const float* __restrict__ th1b, const float* __restrict__ th2w,
    const float* __restrict__ th2b, const float* __restrict__ ab,
    const float* __restrict__ vlw, const float* __restrict__ vlb,
    const float* __restrict__ vls, const float* __restrict__ vlo,
    f16* __restrict__ act) {
  __shared__ float raw[H_][16][64];   // 32 KB score scratch
  __shared__ f16 at2[H_][16][64];     // 16 KB f16 attn for second GEMM
  __shared__ float s_t1[64], s_t2[64], s_b1[H_], s_b2[H_], s_ab[H_ * N_];

  const int b = blockIdx.x, tid = threadIdx.x;
  const int wv = tid >> 5;                 // head handled by this wave
  const int lo = tid & 15, hi = (tid >> 4) & 1;

  if (tid < 64) { s_t1[tid] = th1w[tid]; s_t2[tid] = th2w[tid]; }
  if (tid < H_) { s_b1[tid] = th1b[tid]; s_b2[tid] = th2b[tid]; }
  for (int i = tid; i < H_ * N_; i += 256) s_ab[i] = ab[i];
  __syncthreads();

  const f16* qb = qh + ((long)b * H_ + wv) * NP_ * KD_;
  const f16* kb = kh + ((long)b * H_ + wv) * NP_ * KD_;
  const f16* vb = vh + ((long)b * H_ + wv) * D_ * NP_;

  for (int nt = 0; nt < 4; ++nt) {
    // ---- phase 1: raw = q @ k^T * SCALE + rel-pos bias (WMMA, per head) --
    {
      v16h a = load_frag_a(qb + nt * 16 * KD_, KD_);
#pragma unroll
      for (int mt = 0; mt < 4; ++mt) {
        v16h bf = load_frag_b(kb + mt * 16 * KD_, KD_);
        v8f c = {};
        c = wmma16(a, bf, c);
#pragma unroll
        for (int r = 0; r < 8; ++r) {
          const int nl = r + 8 * hi;
          const int ng = nt * 16 + nl;
          const int mg = mt * 16 + lo;
          float bias = 0.f;
          if (ng < N_ && mg < N_) {
            int dy = ng / 7 - mg / 7; if (dy < 0) dy = -dy;
            int dx = ng % 7 - mg % 7; if (dx < 0) dx = -dx;
            bias = s_ab[wv * N_ + dy * 7 + dx];
          }
          raw[wv][nl][mg] = c[r] * SCALE_ + bias;
        }
      }
    }
    __syncthreads();
    // ---- phase 2: talking-heads-1 + column mask (in place per column) ----
    for (int e = tid; e < 16 * 64; e += 256) {
      const int nl = e >> 6, m = e & 63;
      float col[H_];
#pragma unroll
      for (int j = 0; j < H_; ++j) col[j] = raw[j][nl][m];
#pragma unroll
      for (int i = 0; i < H_; ++i) {
        float s = s_b1[i];
#pragma unroll
        for (int j = 0; j < H_; ++j) s += s_t1[i * H_ + j] * col[j];
        raw[i][nl][m] = (m < N_) ? s : -1e30f;
      }
    }
    __syncthreads();
    // ---- phase 3: softmax over m (one thread per (head,row)) ------------
    if (tid < 128) {
      const int i = tid >> 4, nl = tid & 15;
      float mx = -1e30f;
      for (int m = 0; m < 64; ++m) mx = fmaxf(mx, raw[i][nl][m]);
      float sum = 0.f;
      for (int m = 0; m < 64; ++m) sum += expf(raw[i][nl][m] - mx);
      const float inv = 1.f / sum;
      for (int m = 0; m < 64; ++m)
        raw[i][nl][m] = expf(raw[i][nl][m] - mx) * inv;
    }
    __syncthreads();
    // ---- phase 4: talking-heads-2, convert to f16 ------------------------
    for (int e = tid; e < 16 * 64; e += 256) {
      const int nl = e >> 6, m = e & 63;
      float col[H_];
#pragma unroll
      for (int j = 0; j < H_; ++j) col[j] = raw[j][nl][m];
#pragma unroll
      for (int i = 0; i < H_; ++i) {
        float s = s_b2[i];
#pragma unroll
        for (int j = 0; j < H_; ++j) s += s_t2[i * H_ + j] * col[j];
        at2[i][nl][m] = (f16)s;
      }
    }
    __syncthreads();
    // ---- phase 5: ctx = attn @ v (WMMA) + depthwise/BN/GELU epilogue -----
    const f16* a2 = (const f16*)&at2[wv][0][0];   // 16x64, ld=64
    for (int dt = 0; dt < 8; ++dt) {
      v8f c = {};
#pragma unroll
      for (int ks = 0; ks < 2; ++ks) {
        v16h a = load_frag_a(a2 + ks * 32, 64);
        v16h bf = load_frag_b(vb + (dt * 16) * NP_ + ks * 32, NP_);
        c = wmma16(a, bf, c);
      }
      const int dg = dt * 16 + lo;
      const int ch = wv * D_ + dg;
#pragma unroll
      for (int r = 0; r < 8; ++r) {
        const int ng = nt * 16 + r + 8 * hi;
        if (ng >= N_) continue;
        const int y = ng / 7, x = ng % 7;
        float conv = 0.f;
#pragma unroll
        for (int dy = -1; dy <= 1; ++dy) {
#pragma unroll
          for (int dx = -1; dx <= 1; ++dx) {
            const int yy = y + dy, xx = x + dx;
            if (yy < 0 || yy > 6 || xx < 0 || xx > 6) continue;
            conv += (float)vb[dg * NP_ + yy * 7 + xx] *
                    vlw[ch * 9 + (dy + 1) * 3 + (dx + 1)];
          }
        }
        const float vl = (conv + vlb[ch]) * vls[ch] + vlo[ch];
        const float o = c[r] + vl;
        const float g = 0.5f * o * (1.f + erff(o * 0.70710678118654752f));
        act[((long)b * N_ + ng) * DH_ + ch] = (f16)g;
      }
    }
    __syncthreads();
  }
}

// ---------------- K3: projection GEMM (384 x 1024 x 50176) ----------------
// Block-cooperative: one block per 64-column activation panel; the panel's
// 64x32 f16 B-slab is staged into LDS with async tensor-path copies
// (global_load_async_to_lds_b128, ASYNCcnt) and double-buffered. 8 waves x
// 3 M-tiles each cover all 24 output-channel tiles, so B is loaded from
// global exactly once per panel per K-step.
__global__ void __launch_bounds__(256) p_gemm(const f16* __restrict__ pwh,
                                              const f16* __restrict__ acth,
                                              const float* __restrict__ padd,
                                              float* __restrict__ out) {
  __shared__ f16 bstage[2][64 * 32];  // 2 x 4 KB
  const int tid = threadIdx.x;
  const int wv = tid >> 5;
  const int n0 = blockIdx.x * 64;
  const int row = tid >> 2, kp = (tid & 3) * 8;  // 256 thr x 16B = 4 KB slab

  auto stage = [&](int ks, int buf) {
    const f16* g = acth + (long)(n0 + row) * DH_ + ks * 32 + kp;
    unsigned lofs = (unsigned)(size_t)&bstage[buf][row * 32 + kp];
    asm volatile("global_load_async_to_lds_b128 %0, %1, off"
                 :: "v"(lofs), "v"(g) : "memory");
  };

  stage(0, 0);
  asm volatile("s_wait_asynccnt 0x0" ::: "memory");
  __syncthreads();

  v8f acc[3][4] = {};
  for (int ks = 0; ks < DH_ / 32; ++ks) {
    const int cur = ks & 1;
    if (ks + 1 < DH_ / 32) stage(ks + 1, cur ^ 1);
    const f16* bb = (const f16*)&bstage[cur][0];
#pragma unroll
    for (int t = 0; t < 3; ++t) {
      const int m0 = (wv + 8 * t) * 16;
      v16h a = load_frag_a(pwh + (long)m0 * DH_ + ks * 32, DH_);
#pragma unroll
      for (int s = 0; s < 4; ++s) {
        v16h bf = load_frag_b(bb + (s * 16) * 32, 32);
        acc[t][s] = wmma16(a, bf, acc[t][s]);
      }
    }
    if (ks + 1 < DH_ / 32) {
      asm volatile("s_wait_asynccnt 0x0" ::: "memory");
      __syncthreads();
    }
  }

  const int lo = tid & 15, hi = (tid >> 4) & 1;
#pragma unroll
  for (int t = 0; t < 3; ++t) {
#pragma unroll
    for (int s = 0; s < 4; ++s) {
#pragma unroll
      for (int r = 0; r < 8; ++r) {
        const int o = (wv + 8 * t) * 16 + r + 8 * hi;
        const int mc = n0 + s * 16 + lo;
        const int b = mc / N_, n = mc % N_;
        out[((long)b * DIM_ + o) * N_ + n] = acc[t][s][r] + padd[o];
      }
    }
  }
}

// ---------------------------- launch ----------------------------
extern "C" void kernel_launch(void* const* d_in, const int* in_sizes, int n_in,
                              void* d_out, int out_size, void* d_ws,
                              size_t ws_size, hipStream_t stream) {
  const float* x    = (const float*)d_in[0];
  const float* q_w  = (const float*)d_in[1];
  const float* q_b  = (const float*)d_in[2];
  const float* q_s  = (const float*)d_in[3];
  const float* q_o  = (const float*)d_in[4];
  const float* k_w  = (const float*)d_in[5];
  const float* k_b  = (const float*)d_in[6];
  const float* k_s  = (const float*)d_in[7];
  const float* k_o  = (const float*)d_in[8];
  const float* v_w  = (const float*)d_in[9];
  const float* v_b  = (const float*)d_in[10];
  const float* v_s  = (const float*)d_in[11];
  const float* v_o  = (const float*)d_in[12];
  const float* vl_w = (const float*)d_in[13];
  const float* vl_b = (const float*)d_in[14];
  const float* vl_s = (const float*)d_in[15];
  const float* vl_o = (const float*)d_in[16];
  const float* th1w = (const float*)d_in[17];
  const float* th1b = (const float*)d_in[18];
  const float* th2w = (const float*)d_in[19];
  const float* th2b = (const float*)d_in[20];
  const float* p_w  = (const float*)d_in[21];
  const float* p_b  = (const float*)d_in[22];
  const float* p_s  = (const float*)d_in[23];
  const float* p_o  = (const float*)d_in[24];
  const float* ab   = (const float*)d_in[25];
  float* out = (float*)d_out;

  char* ws = (char*)d_ws;
  size_t off = 0;
  auto carve = [&](size_t bytes) -> void* {
    void* p = ws + off;
    off += (bytes + 255) & ~(size_t)255;
    return p;
  };
  f16*   wqkv   = (f16*)carve((size_t)OC_ * DIM_ * 2);
  f16*   pwh    = (f16*)carve((size_t)DIM_ * DH_ * 2);
  float* qkvadd = (float*)carve((size_t)OC_ * 4);
  float* padd   = (float*)carve((size_t)DIM_ * 4);
  f16*   xt     = (f16*)carve((size_t)MCOL_ * DIM_ * 2);
  f16*   qhb    = (f16*)carve((size_t)B_ * H_ * NP_ * KD_ * 2);
  f16*   khb    = (f16*)carve((size_t)B_ * H_ * NP_ * KD_ * 2);
  f16*   vhb    = (f16*)carve((size_t)B_ * H_ * D_ * NP_ * 2);
  f16*   acth   = (f16*)carve((size_t)MCOL_ * DH_ * 2);
  (void)ws_size; (void)n_in; (void)in_sizes; (void)out_size;

  {
    const long total = (long)OC_ * DIM_ + (long)DIM_ * DH_ + OC_ + DIM_;
    pack_weights<<<(unsigned)((total + 255) / 256), 256, 0, stream>>>(
        q_w, q_b, q_s, q_o, k_w, k_b, k_s, k_o, v_w, v_b, v_s, v_o,
        p_w, p_b, p_s, p_o, wqkv, qkvadd, pwh, padd);
  }
  {
    const long total = (long)MCOL_ * DIM_;
    transpose_x<<<(unsigned)((total + 255) / 256), 256, 0, stream>>>(x, xt);
  }
  {
    const long bytes = ((long)B_ * H_ * NP_ * KD_ * 2) * 2 +
                       (long)B_ * H_ * D_ * NP_ * 2;
    zero_u32<<<4096, 256, 0, stream>>>((unsigned int*)qhb, bytes / 4);
  }
  qkv_gemm<<<(96 * NT4_) / 8, 256, 0, stream>>>(wqkv, xt, qkvadd, qhb, khb, vhb);
  attn_kernel<<<B_, 256, 0, stream>>>(qhb, khb, vhb, th1w, th1b, th2w, th2b,
                                      ab, vl_w, vl_b, vl_s, vl_o, acth);
  p_gemm<<<NT4_, 256, 0, stream>>>(pwh, acth, padd, out);
}
